// QuantizeInterpolatedEMAResetAttention_63866163692088
// MI455X (gfx1250) — matrix operand order repaired and checked by reading, not compiled, not run on hardware
//
#include <hip/hip_runtime.h>
#include <hip/hip_bf16.h>

// ---------------- problem constants ----------------
constexpr int NB = 4;          // batch
constexpr int NC = 512;        // channels
constexpr int NT = 1024;       // time (full)
constexpr int NQ = 256;        // pooled time
constexpr int NV = 4096;       // codebook size
constexpr int NH = 8;          // heads
constexpr int ND = 64;         // head dim
constexpr int MQ   = NB * NQ;      // 1024 pooled tokens
constexpr int MF   = NB * NT;      // 4096 full tokens
constexpr int MTOT = MQ + MF;      // 5120 tokens total
constexpr float EPSV = 1e-5f;
constexpr float CSCL = 0.044194173824159216f; // 1/(sqrt(64)*sqrt(8))

typedef __attribute__((ext_vector_type(16))) _Float16 v16h;
typedef __attribute__((ext_vector_type(8)))  _Float16 v8h;
typedef __attribute__((ext_vector_type(8)))  float    v8f;

// ---------------- workspace layout (bytes, every chunk 256B-aligned) -------
constexpr size_t OFF_HS   = 0;                                    // MTOT*NC f32
constexpr size_t OFF_HSH  = OFF_HS   + (size_t)MTOT * NC * 4;     // MTOT*NC f16
constexpr size_t OFF_WQT  = OFF_HSH  + (size_t)MTOT * NC * 2;     // NC*NC f16 (W^T)
constexpr size_t OFF_WKT  = OFF_WQT  + (size_t)NC * NC * 2;
constexpr size_t OFF_WVT  = OFF_WKT  + (size_t)NC * NC * 2;
constexpr size_t OFF_CBH  = OFF_WVT  + (size_t)NC * NC * 2;       // NV*NC f16
constexpr size_t OFF_QRAW = OFF_CBH  + (size_t)NV * NC * 2;       // MTOT*NC f32
constexpr size_t OFF_KRAW = OFF_QRAW + (size_t)MTOT * NC * 4;     // NV*NC f32
constexpr size_t OFF_VAL  = OFF_KRAW + (size_t)NV * NC * 4;       // NV*NC f32
constexpr size_t OFF_QTH  = OFF_VAL  + (size_t)NV * NC * 4;       // MTOT*NC f16
constexpr size_t OFF_KNH  = OFF_QTH  + (size_t)MTOT * NC * 2;     // NV*NC f16
constexpr size_t OFF_C    = OFF_KNH  + (size_t)NV * NC * 2;       // MTOT*NH f32
constexpr size_t OFF_IDXQ = OFF_C    + (size_t)MTOT * NH * 4;     // MQ i32
constexpr size_t OFF_WQV  = OFF_IDXQ + (size_t)MQ * 4;            // MQ f32
constexpr size_t OFF_IDXF = OFF_WQV  + (size_t)MQ * 4;            // MF i32
constexpr size_t OFF_CNT  = OFF_IDXF + (size_t)MF * 4;            // NV f32

// ================= K0: pool + transpose, f32 + f16 copies ==================
__global__ void k_build_hs(const float* __restrict__ z,
                           float* __restrict__ hs, _Float16* __restrict__ hsh) {
  int id = blockIdx.x * blockDim.x + threadIdx.x;
  if (id >= MTOT * NC) return;
  int r = id / NC, c = id % NC;
  float v;
  if (r < MQ) {                              // pooled tokens: mean over 4
    int b = r / NQ, tq = r % NQ;
    const float* p = z + (size_t)b * NC * NT + (size_t)c * NT + tq * 4;
    v = 0.25f * (p[0] + p[1] + p[2] + p[3]);
  } else {                                   // full tokens: transpose
    int rr = r - MQ; int b = rr / NT; int t = rr % NT;
    v = z[(size_t)b * NC * NT + (size_t)c * NT + t];
  }
  hs[id] = v;
  hsh[id] = (_Float16)v;
}

// ======= K1: W^T -> f16 for Wq/Wk/Wv, codebook -> f16 ======================
__global__ void k_convert(const float* __restrict__ Wq, const float* __restrict__ Wk,
                          const float* __restrict__ Wv, const float* __restrict__ cb,
                          _Float16* __restrict__ WqT, _Float16* __restrict__ WkT,
                          _Float16* __restrict__ WvT, _Float16* __restrict__ cbH) {
  int id = blockIdx.x * blockDim.x + threadIdx.x;
  constexpr int WTOT = NC * NC;              // 262144
  if (id < 3 * WTOT) {
    int w = id / WTOT, rem = id % WTOT;
    int n = rem / NC, k = rem % NC;
    const float* W  = (w == 0) ? Wq  : (w == 1) ? Wk  : Wv;
    _Float16*    Ot = (w == 0) ? WqT : (w == 1) ? WkT : WvT;
    Ot[n * NC + k] = (_Float16)W[k * NC + n];   // transpose
  } else {
    int i = id - 3 * WTOT;
    if (i < NV * NC) cbH[i] = (_Float16)cb[i];
  }
}

// ======= K2: generic GEMM  Out(MxN,f32) = A(MxK,f16) * Bt(NxK,f16)^T + bias =
// block = 256 threads = 8 waves; wave -> one 16x16 tile; grid = (N/128, M/16)
__global__ void k_gemm_nt(const _Float16* __restrict__ A, const _Float16* __restrict__ Bt,
                          const float* __restrict__ bias, float* __restrict__ Out,
                          int M, int N, int K) {
  int wave = threadIdx.x >> 5;
  int lane = threadIdx.x & 31;
  int hi   = lane >> 4;
  int nBase = blockIdx.x * 128 + wave * 16;
  int mBase = blockIdx.y * 16;
  int mrow = mBase + (lane & 15);
  int nrow = nBase + (lane & 15);
  const _Float16* pa = A  + (size_t)mrow * K + hi * 16;
  const _Float16* pb = Bt + (size_t)nrow * K + hi * 16;
  v8f acc = {};
  for (int kb = 0; kb < K; kb += 32) {
    v8h alo = *(const v8h*)(pa + kb);
    v8h ahi = *(const v8h*)(pa + kb + 8);
    v16h af;
#pragma unroll
    for (int i = 0; i < 8; ++i) { af[i] = alo[i]; af[8 + i] = ahi[i]; }
    v16h bf = *(const v16h*)(pb + kb);
    acc = __builtin_amdgcn_wmma_f32_16x16x32_f16(false, af, false, bf,
                                                 (short)0, acc, false, false);
  }
  int n = nBase + (lane & 15);
  float bv = bias[n];
#pragma unroll
  for (int e = 0; e < 8; ++e) {
    int m = mBase + e + hi * 8;
    Out[(size_t)m * N + n] = acc[e] + bv;
  }
}

// ======= K3: head-pool weights c = hs @ Wp + bp  (N=8, plain VALU) =========
__global__ void k_pool_c(const float* __restrict__ hs, const float* __restrict__ Wp,
                         const float* __restrict__ bp, float* __restrict__ cvec) {
  int id = blockIdx.x * blockDim.x + threadIdx.x;
  if (id >= MTOT * NH) return;
  int r = id / NH, h = id % NH;
  const float* x = hs + (size_t)r * NC;
  float s = bp[h];
  for (int k = 0; k < NC; ++k) s += x[k] * Wp[k * NH + h];
  cvec[id] = s;
}

// ======= K4: per-head RMSNorm + scale folding -> f16 GEMM operands =========
// one wave per 64-elem head segment; rows [0,MTOT) are Q, [MTOT,MTOT+NV) are K
__global__ void k_norm(const float* __restrict__ Qraw, const float* __restrict__ Kraw,
                       const float* __restrict__ cvec,
                       const float* __restrict__ gq, const float* __restrict__ gk,
                       _Float16* __restrict__ QtH, _Float16* __restrict__ KnH) {
  int wave = threadIdx.x >> 5, lane = threadIdx.x & 31;
  int seg = blockIdx.x * 8 + wave;
  int row = seg / NH, h = seg % NH;
  bool isQ = row < MTOT;
  const float* src = isQ ? (Qraw + (size_t)row * NC + h * ND)
                         : (Kraw + (size_t)(row - MTOT) * NC + h * ND);
  float x0 = src[lane], x1 = src[lane + 32];
  float ss = x0 * x0 + x1 * x1;
#pragma unroll
  for (int m = 16; m > 0; m >>= 1) ss += __shfl_xor(ss, m, 32);
  float scale = rsqrtf(ss * (1.0f / (float)ND) + EPSV);
  if (isQ) {
    float cm = cvec[row * NH + h] * CSCL;        // fold c / (sqrt(D)*sqrt(H))
    _Float16* dst = QtH + (size_t)row * NC + h * ND;
    dst[lane]      = (_Float16)(x0 * scale * gq[lane]      * cm);
    dst[lane + 32] = (_Float16)(x1 * scale * gq[lane + 32] * cm);
  } else {
    _Float16* dst = KnH + (size_t)(row - MTOT) * NC + h * ND;
    dst[lane]      = (_Float16)(x0 * scale * gk[lane]);
    dst[lane + 32] = (_Float16)(x1 * scale * gk[lane + 32]);
  }
}

// ======= K5: fused logits GEMM + online softmax stats ======================
// block = 256 thr (8 waves) handles 16 token rows; loops V in chunks of 512;
// wave computes 4 N-tiles with a shared A fragment (K-outer, 4 accumulators);
// logits chunk kept in LDS; running (max,argmax,rescaled sumexp) per row.
constexpr int NCHUNK = 512;
__global__ void k_logits(const _Float16* __restrict__ QtH, const _Float16* __restrict__ KnH,
                         int* __restrict__ idx_q, float* __restrict__ w_q,
                         int* __restrict__ idx_f) {
  __shared__ float tileL[16][NCHUNK];
  __shared__ float partMax[16][16];
  __shared__ int   partIdx[16][16];
  __shared__ float partSum[16][16];
  __shared__ float runMax[16];
  __shared__ int   runIdx[16];
  __shared__ float runSum[16];

  int tid  = threadIdx.x;
  int wave = tid >> 5, lane = tid & 31;
  int hi   = lane >> 4;
  int nl   = lane & 15;
  int mBase = blockIdx.x * 16;
  int mrow  = mBase + nl;
  if (tid < 16) { runMax[tid] = -3.4e38f; runIdx[tid] = 0; runSum[tid] = 0.0f; }
  __syncthreads();

  const _Float16* pa0 = QtH + (size_t)mrow * NC + hi * 16;

  for (int chunk = 0; chunk < NV / NCHUNK; ++chunk) {
    int n0 = chunk * NCHUNK;
    // --- GEMM: wave computes 4 consecutive 16x16 tiles of this chunk ------
    {
      const _Float16* pb0 = KnH + (size_t)(n0 + wave * 64 + nl) * NC + hi * 16;
      const _Float16* pb1 = pb0 + (size_t)16 * NC;
      const _Float16* pb2 = pb0 + (size_t)32 * NC;
      const _Float16* pb3 = pb0 + (size_t)48 * NC;
      // prefetch next chunk's B panel rows into cache while computing
      if (chunk + 1 < NV / NCHUNK) {
        __builtin_prefetch(pb0 + (size_t)NCHUNK * NC, 0, 1);
        __builtin_prefetch(pb1 + (size_t)NCHUNK * NC, 0, 1);
        __builtin_prefetch(pb2 + (size_t)NCHUNK * NC, 0, 1);
        __builtin_prefetch(pb3 + (size_t)NCHUNK * NC, 0, 1);
      }
      v8f acc0 = {}, acc1 = {}, acc2 = {}, acc3 = {};
      for (int kb = 0; kb < NC; kb += 32) {
        v8h alo = *(const v8h*)(pa0 + kb);
        v8h ahi = *(const v8h*)(pa0 + kb + 8);
        v16h af;
#pragma unroll
        for (int i = 0; i < 8; ++i) { af[i] = alo[i]; af[8 + i] = ahi[i]; }
        v16h b0 = *(const v16h*)(pb0 + kb);
        v16h b1 = *(const v16h*)(pb1 + kb);
        v16h b2 = *(const v16h*)(pb2 + kb);
        v16h b3 = *(const v16h*)(pb3 + kb);
        acc0 = __builtin_amdgcn_wmma_f32_16x16x32_f16(false, af, false, b0,
                                                      (short)0, acc0, false, false);
        acc1 = __builtin_amdgcn_wmma_f32_16x16x32_f16(false, af, false, b1,
                                                      (short)0, acc1, false, false);
        acc2 = __builtin_amdgcn_wmma_f32_16x16x32_f16(false, af, false, b2,
                                                      (short)0, acc2, false, false);
        acc3 = __builtin_amdgcn_wmma_f32_16x16x32_f16(false, af, false, b3,
                                                      (short)0, acc3, false, false);
      }
      int col = wave * 64 + nl;
#pragma unroll
      for (int e = 0; e < 8; ++e) {
        int rr = e + hi * 8;
        tileL[rr][col]      = acc0[e];
        tileL[rr][col + 16] = acc1[e];
        tileL[rr][col + 32] = acc2[e];
        tileL[rr][col + 48] = acc3[e];
      }
    }
    __syncthreads();

    // --- stats: 16 threads per row, 32 cols each ---
    int rowm = tid >> 4;
    int sub  = tid & 15;
    {
      float lmax = -3.4e38f; int lidx = 0;
      int c0 = sub * (NCHUNK / 16);
      for (int i = 0; i < NCHUNK / 16; ++i) {
        float v = tileL[rowm][c0 + i];
        if (v > lmax) { lmax = v; lidx = n0 + c0 + i; }
      }
      partMax[rowm][sub] = lmax; partIdx[rowm][sub] = lidx;
    }
    __syncthreads();
    if (tid < 16) {       // combine in increasing-n order -> first-occurrence argmax
      float cmax = partMax[tid][0]; int cidx = partIdx[tid][0];
      for (int i = 1; i < 16; ++i)
        if (partMax[tid][i] > cmax) { cmax = partMax[tid][i]; cidx = partIdx[tid][i]; }
      if (cmax > runMax[tid]) {
        runSum[tid] *= expf(runMax[tid] - cmax);
        runMax[tid] = cmax; runIdx[tid] = cidx;
      }
    }
    __syncthreads();
    {
      float m = runMax[rowm];
      float s = 0.0f;
      int c0 = sub * (NCHUNK / 16);
      for (int i = 0; i < NCHUNK / 16; ++i)
        s += expf(tileL[rowm][c0 + i] - m);
      partSum[rowm][sub] = s;
    }
    __syncthreads();
    if (tid < 16) {
      float s = runSum[tid];
      for (int i = 0; i < 16; ++i) s += partSum[tid][i];
      runSum[tid] = s;
    }
    __syncthreads();
  }

  if (tid < 16) {
    int r = mBase + tid;
    if (r < MQ) {                       // pooled token: straight-through weight
      float s = 1.0f / runSum[tid];
      w_q[r]  = (1.0f - s) + s;
      idx_q[r] = runIdx[tid];
    } else {                            // full-res token: index only
      idx_f[r - MQ] = runIdx[tid];
    }
  }
}

// ======= K6: z_hat = linear-interp( w * value[idx] ), (B,C,T) ==============
__global__ void k_zhat(const int* __restrict__ idx_q, const float* __restrict__ w_q,
                       const float* __restrict__ Val, float* __restrict__ out) {
  int id = blockIdx.x * blockDim.x + threadIdx.x;
  if (id >= NB * NC * NT) return;
  int t = id % NT; int rem = id / NT; int c = rem % NC; int b = rem / NC;
  float pos = (t + 0.5f) * ((float)NQ / (float)NT) - 0.5f;
  pos = fminf(fmaxf(pos, 0.0f), (float)(NQ - 1));
  int i0 = (int)floorf(pos);
  int i1 = (i0 + 1 < NQ) ? i0 + 1 : NQ - 1;
  float w = pos - (float)i0;
  int r0 = b * NQ + i0, r1 = b * NQ + i1;
  float v0 = w_q[r0] * Val[(size_t)idx_q[r0] * NC + c];
  float v1 = w_q[r1] * Val[(size_t)idx_q[r1] * NC + c];
  out[id] = v0 * (1.0f - w) + v1 * w;
}

// ======= K7: histogram + perplexity ========================================
__global__ void k_zero(float* __restrict__ counts) {
  int i = blockIdx.x * blockDim.x + threadIdx.x;
  if (i < NV) counts[i] = 0.0f;
}
__global__ void k_hist(const int* __restrict__ idx_f, float* __restrict__ counts) {
  int i = blockIdx.x * blockDim.x + threadIdx.x;
  if (i < MF) atomicAdd(&counts[idx_f[i]], 1.0f);
}
__global__ void k_perp(const float* __restrict__ counts, float* __restrict__ out) {
  __shared__ float red[256];
  int tid = threadIdx.x;
  const float invTot = 1.0f / (float)MF;      // counts sum == MF exactly
  float e = 0.0f;
  for (int i = tid; i < NV; i += 256) {
    float p = counts[i] * invTot;
    e -= p * logf(p + 1e-7f);
  }
  red[tid] = e; __syncthreads();
#pragma unroll
  for (int s = 128; s > 0; s >>= 1) {
    if (tid < s) red[tid] += red[tid + s];
    __syncthreads();
  }
  if (tid == 0) out[(size_t)NB * NC * NT] = expf(red[0]);
}

// =========================== launcher ======================================
extern "C" void kernel_launch(void* const* d_in, const int* in_sizes, int n_in,
                              void* d_out, int out_size, void* d_ws, size_t ws_size,
                              hipStream_t stream) {
  (void)in_sizes; (void)n_in; (void)out_size; (void)ws_size;
  const float* z  = (const float*)d_in[0];
  // d_in[1] = q (always 256, hardcoded)
  const float* cb = (const float*)d_in[2];
  const float* Wq = (const float*)d_in[3];
  const float* bq = (const float*)d_in[4];
  const float* Wk = (const float*)d_in[5];
  const float* bk = (const float*)d_in[6];
  const float* Wv = (const float*)d_in[7];
  const float* bv = (const float*)d_in[8];
  const float* Wp = (const float*)d_in[9];
  const float* bp = (const float*)d_in[10];
  const float* gq = (const float*)d_in[11];
  const float* gk = (const float*)d_in[12];
  float* out = (float*)d_out;

  char* ws = (char*)d_ws;
  float*    hs   = (float*)(ws + OFF_HS);
  _Float16* hsh  = (_Float16*)(ws + OFF_HSH);
  _Float16* WqT  = (_Float16*)(ws + OFF_WQT);
  _Float16* WkT  = (_Float16*)(ws + OFF_WKT);
  _Float16* WvT  = (_Float16*)(ws + OFF_WVT);
  _Float16* cbH  = (_Float16*)(ws + OFF_CBH);
  float*    Qraw = (float*)(ws + OFF_QRAW);
  float*    Kraw = (float*)(ws + OFF_KRAW);
  float*    Val  = (float*)(ws + OFF_VAL);
  _Float16* QtH  = (_Float16*)(ws + OFF_QTH);
  _Float16* KnH  = (_Float16*)(ws + OFF_KNH);
  float*    cvec = (float*)(ws + OFF_C);
  int*      idxq = (int*)(ws + OFF_IDXQ);
  float*    wqv  = (float*)(ws + OFF_WQV);
  int*      idxf = (int*)(ws + OFF_IDXF);
  float*    cnts = (float*)(ws + OFF_CNT);

  // K0: pooled + full token matrix (f32 and f16)
  k_build_hs<<<(MTOT * NC + 255) / 256, 256, 0, stream>>>(z, hs, hsh);
  // K1: weight transposes + codebook to f16
  k_convert<<<(3 * NC * NC + NV * NC + 255) / 256, 256, 0, stream>>>(
      Wq, Wk, Wv, cb, WqT, WkT, WvT, cbH);
  // K2: projection GEMMs (WMMA)
  k_gemm_nt<<<dim3(NC / 128, MTOT / 16), 256, 0, stream>>>(hsh, WqT, bq, Qraw, MTOT, NC, NC);
  k_gemm_nt<<<dim3(NC / 128, NV / 16),   256, 0, stream>>>(cbH, WkT, bk, Kraw, NV,   NC, NC);
  k_gemm_nt<<<dim3(NC / 128, NV / 16),   256, 0, stream>>>(cbH, WvT, bv, Val,  NV,   NC, NC);
  // K3: head-pool weights
  k_pool_c<<<(MTOT * NH + 255) / 256, 256, 0, stream>>>(hs, Wp, bp, cvec);
  // K4: RMSNorm + scale fold -> f16 operands
  k_norm<<<((MTOT + NV) * NH) / 8, 256, 0, stream>>>(Qraw, Kraw, cvec, gq, gk, QtH, KnH);
  // K5: fused logits GEMM + softmax stats (WMMA)
  k_logits<<<MTOT / 16, 256, 0, stream>>>(QtH, KnH, idxq, wqv, idxf);
  // K6: z_hat gather + interp
  k_zhat<<<(NB * NC * NT + 255) / 256, 256, 0, stream>>>(idxq, wqv, Val, out);
  // K7: perplexity
  k_zero<<<(NV + 255) / 256, 256, 0, stream>>>(cnts);
  k_hist<<<(MF + 255) / 256, 256, 0, stream>>>(idxf, cnts);
  k_perp<<<1, 256, 0, stream>>>(cnts, out);
}
